// NLSALayer_6373731467572
// MI455X (gfx1250) — matrix operand order repaired and checked by reading, not compiled, not run on hardware
//
#include <hip/hip_runtime.h>
#include <hip/hip_bf16.h>

typedef __attribute__((ext_vector_type(16))) _Float16 v16h;
typedef __attribute__((ext_vector_type(8)))  _Float16 v8h;
typedef __attribute__((ext_vector_type(8)))  float    v8f;

#define L_TOT 65536
#define CHUNK_SZ 144
#define KC 456      // chunks per hash round (65664 / 144)
#define PADN 128    // -L % 144
#define NBKT 128
#define NHH 4
#define DD 16
#define HH 64
#define WW 64

__device__ __forceinline__ v8f wmma_f16(v16h a, v16h b, v8f c) {
  return __builtin_amdgcn_wmma_f32_16x16x32_f16(false, a, false, b, (short)0, c, false, false);
}

// A fragment from channel-minor activations: base = act + pos*64 + cb.
// Lane layout (16-bit A 16x32): elements 0..15 -> k = hi*16+e (matches B fragments).
// => two contiguous 16B loads, zeroed by one vector select when out of bounds.
__device__ __forceinline__ v16h loadA64(const _Float16* __restrict__ base, int hi, bool ok) {
  v8h l = *(const v8h*)(base + hi * 16);
  v8h u = *(const v8h*)(base + hi * 16 + 8);
  v8h z = {};
  if (!ok) { l = z; u = z; }
  return __builtin_shufflevector(l, u, 0, 1, 2, 3, 4, 5, 6, 7, 8, 9, 10, 11, 12, 13, 14, 15);
}

// B fragment from fragment-native weights: 32 contiguous halfs per (blk, n);
// lane takes halfs [hi*16 .. hi*16+15] => one contiguous 32B load.
__device__ __forceinline__ v16h loadB32(const _Float16* __restrict__ base, int hi) {
  return *(const v16h*)(base + hi * 16);
}

// ---------------- K0: precision/layout prep -------------------------------
// xh  : [t][cin]                         (channel-minor activations)
// wmB : ((tap*2+cbi)*16 + n)*32 + k      (B-fragment native)
// waB : ((cbi*4+nt)*16 + n)*32 + k
// wcB : (((tap*2+cbi)*4+nt)*16 + n)*32 + k
__global__ void __launch_bounds__(256) k_prep(const float* __restrict__ x,
    const float* __restrict__ wm, const float* __restrict__ wa, const float* __restrict__ wc,
    _Float16* __restrict__ xh, _Float16* __restrict__ wmB,
    _Float16* __restrict__ waB, _Float16* __restrict__ wcB) {
  long i = (long)blockIdx.x * 256 + threadIdx.x;
  if (i < 64L * L_TOT) {
    int c = (int)(i >> 16); int t = (int)(i & 65535);
    xh[(size_t)t * 64 + c] = (_Float16)x[i];
  }
  if (i < 27L * 2 * 16 * 32) {
    int k = (int)(i & 31); int n = (int)((i >> 5) & 15);
    int cbi = (int)((i >> 9) & 1); int tap = (int)(i >> 10);
    wmB[i] = (_Float16)wm[(n * 64 + cbi * 32 + k) * 27 + tap];
  }
  if (i < 2L * 4 * 16 * 32) {
    int k = (int)(i & 31); int n = (int)((i >> 5) & 15);
    int nt = (int)((i >> 9) & 3); int cbi = (int)(i >> 11);
    waB[i] = (_Float16)wa[(nt * 16 + n) * 64 + cbi * 32 + k];
  }
  if (i < 27L * 2 * 4 * 16 * 32) {
    int k = (int)(i & 31); int n = (int)((i >> 5) & 15);
    int nt = (int)((i >> 9) & 3); int cbi = (int)((i >> 11) & 1); int tap = (int)(i >> 12);
    wcB[i] = (_Float16)wc[((nt * 16 + n) * 64 + cbi * 32 + k) * 27 + tap];
  }
}

// ---------------- K1: match conv 64->16, 3x3x3, pad 1 (implicit GEMM) -----
// One wave handles 32 consecutive positions (2 M-tiles) -> B fragments reused x2.
__global__ void __launch_bounds__(32) k_conv_match(const _Float16* __restrict__ xh,
    const _Float16* __restrict__ wmB, const float* __restrict__ bm, float* __restrict__ xe) {
  int tbase = blockIdx.x * 32;
  int lane = threadIdx.x; int m = lane & 15; int hi = lane >> 4;
  int pos0 = tbase + m;           // same (d,h) row for both tiles
  int d = pos0 >> 12; int h = (pos0 >> 6) & 63; int w0 = pos0 & 63;
  v8f acc[2] = {{}, {}};
  for (int tap = 0; tap < 27; ++tap) {
    int dz = tap / 9 - 1, dy = (tap / 3) % 3 - 1, dx = tap % 3 - 1;
    bool okdh = (unsigned)(d + dz) < DD && (unsigned)(h + dy) < HH;
    bool ok0 = okdh && (unsigned)(w0 + dx) < WW;
    bool ok1 = okdh && (unsigned)(w0 + 16 + dx) < WW;
    int off = dz * 4096 + dy * 64 + dx;
    int p0 = ok0 ? (pos0 + off) : pos0;
    int p1 = ok1 ? (pos0 + 16 + off) : (pos0 + 16);
    if (tap < 26) __builtin_prefetch(wmB + (size_t)(tap + 1) * 1024, 0, 1);
    for (int cbi = 0; cbi < 2; ++cbi) {
      v16h b = loadB32(wmB + (size_t)((tap * 2 + cbi) * 16 + m) * 32, hi);
      v16h a0 = loadA64(xh + (size_t)p0 * 64 + cbi * 32, hi, ok0);
      acc[0] = wmma_f16(a0, b, acc[0]);
      v16h a1 = loadA64(xh + (size_t)p1 * 64 + cbi * 32, hi, ok1);
      acc[1] = wmma_f16(a1, b, acc[1]);
    }
  }
  for (int mt = 0; mt < 2; ++mt)
    for (int r = 0; r < 8; ++r) {
      int row = r + 8 * hi;
      xe[(size_t)(tbase + mt * 16 + row) * 16 + m] = acc[mt][r] + bm[m];
    }
}

// ---------------- K2: assembly conv 1x1x1 64->64 --------------------------
__global__ void __launch_bounds__(32) k_conv_assembly(const _Float16* __restrict__ xh,
    const _Float16* __restrict__ waB, const float* __restrict__ ba, _Float16* __restrict__ yef) {
  int tbase = blockIdx.x * 16;
  int lane = threadIdx.x; int m = lane & 15; int hi = lane >> 4;
  v8f acc[4] = {{}, {}, {}, {}};
  const _Float16* abase = xh + (size_t)(tbase + m) * 64;
  for (int cbi = 0; cbi < 2; ++cbi) {
    v16h a = loadA64(abase + cbi * 32, hi, true);
    for (int nt = 0; nt < 4; ++nt) {
      v16h b = loadB32(waB + (size_t)((cbi * 4 + nt) * 16 + m) * 32, hi);
      acc[nt] = wmma_f16(a, b, acc[nt]);
    }
  }
  for (int nt = 0; nt < 4; ++nt)
    for (int r = 0; r < 8; ++r) {
      int row = r + 8 * hi; int co = nt * 16 + m;
      yef[(size_t)(tbase + row) * 64 + co] = (_Float16)(acc[nt][r] + ba[co]);
    }
}

// ---------------- K3: LSH hash codes --------------------------------------
__global__ void __launch_bounds__(256) k_hash(const float* __restrict__ xe,
    const float* __restrict__ rot, int* __restrict__ codes) {
  __shared__ float rl[16 * 256];
  for (int i = threadIdx.x; i < 4096; i += 256) rl[i] = rot[i];
  __syncthreads();
  int t = blockIdx.x * 256 + threadIdx.x;
  float xr[16];
  for (int c = 0; c < 16; ++c) xr[c] = xe[(size_t)t * 16 + c];
  for (int h = 0; h < NHH; ++h) {
    float bmax = -1e30f, bmin = 1e30f; int im = 0, imn = 0;
    for (int i = 0; i < 64; ++i) {
      float s = 0.f;
      for (int c = 0; c < 16; ++c) s += xr[c] * rl[c * 256 + h * 64 + i];
      if (s > bmax) { bmax = s; im = i; }
      if (s < bmin) { bmin = s; imn = i; }
    }
    codes[h * L_TOT + t] = (bmax >= -bmin) ? im : (64 + imn);
  }
}

// ---------------- K4: stable counting sort (hist / offsets / scatter) -----
__global__ void __launch_bounds__(256) k_hist(const int* __restrict__ codes, int* __restrict__ bh) {
  int h = blockIdx.x >> 6; int sb = blockIdx.x & 63;
  __shared__ int cnt[NBKT];
  if (threadIdx.x < NBKT) cnt[threadIdx.x] = 0;
  __syncthreads();
  int base = h * L_TOT + sb * 1024;
  for (int j = threadIdx.x; j < 1024; j += 256) atomicAdd(&cnt[codes[base + j]], 1);
  __syncthreads();
  if (threadIdx.x < NBKT) bh[(h * 64 + sb) * NBKT + threadIdx.x] = cnt[threadIdx.x];
}

__global__ void __launch_bounds__(128) k_offsets(const int* __restrict__ bh, int* __restrict__ bo) {
  int h = blockIdx.x; int c = threadIdx.x;
  __shared__ int tot[NBKT]; __shared__ int basev[NBKT];
  int run = 0;
  for (int sb = 0; sb < 64; ++sb) {
    bo[(h * 64 + sb) * NBKT + c] = run;
    run += bh[(h * 64 + sb) * NBKT + c];
  }
  tot[c] = run;
  __syncthreads();
  if (c == 0) { int a = 0; for (int i = 0; i < NBKT; ++i) { basev[i] = a; a += tot[i]; } }
  __syncthreads();
  int b0 = basev[c];
  for (int sb = 0; sb < 64; ++sb) bo[(h * 64 + sb) * NBKT + c] += b0;
}

__global__ void __launch_bounds__(1024) k_sort_scatter(const int* __restrict__ codes,
    const int* __restrict__ bo, int* __restrict__ sortedT) {
  int h = blockIdx.x >> 6; int sb = blockIdx.x & 63;
  __shared__ int lc[1024];
  int j = threadIdx.x;
  int t = sb * 1024 + j;
  int code = codes[h * L_TOT + t];
  lc[j] = code;
  __syncthreads();
  int rank = 0;
  for (int jj = 0; jj < j; ++jj) rank += (lc[jj] == code);
  sortedT[h * L_TOT + bo[(h * 64 + sb) * NBKT + code] + rank] = t;
}

// ---------------- K5: chunked attention (flash-style, WMMA) ---------------
// grid: NHH*KC*3 blocks of 96 threads (3 waves, one 16-row M-tile each)
__global__ void __launch_bounds__(96) k_attention(const float* __restrict__ xe,
    const _Float16* __restrict__ yef, const int* __restrict__ sortedT,
    float* __restrict__ retW, float* __restrict__ bsW) {
  __shared__ alignas(32) _Float16 knL[432 * 16];
  __shared__ alignas(32) _Float16 qL[48 * 16];
  __shared__ alignas(32) _Float16 st[3][16][16];
  __shared__ alignas(32) _Float16 vTt[64 * 16];   // V tile transposed: [c][k]
  __shared__ int gT[432];

  int b = blockIdx.x;
  int h = b / (KC * 3);
  int rem = b % (KC * 3);
  int ck = rem / 3;
  int mb = rem % 3;
  int tid = threadIdx.x;
  size_t hbase = (size_t)h * L_TOT;

  // keys: normalized rows of chunks {ck, ck-1, ck+1} (circular, padded tail dup)
  for (int j = tid; j < 432; j += 96) {
    int seg = j / CHUNK_SZ; int e = j % CHUNK_SZ;
    int cc = (seg == 0) ? ck : (seg == 1 ? (ck + KC - 1) % KC : (ck + 1) % KC);
    int pidx = cc * CHUNK_SZ + e;
    int s_eff = pidx < L_TOT ? pidx : pidx - PADN;
    int t = sortedT[hbase + s_eff];
    gT[j] = t;
    float v[16]; float nrm = 0.f;
    for (int c = 0; c < 16; ++c) { v[c] = xe[(size_t)t * 16 + c]; nrm += v[c] * v[c]; }
    float inv = 1.f / fmaxf(sqrtf(nrm), 5e-05f);
    for (int c = 0; c < 16; ++c) knL[j * 16 + c] = (_Float16)(v[c] * inv);
  }
  // queries: this block's 48 rows, unnormalized
  for (int i = tid; i < 48; i += 96) {
    int row = mb * 48 + i;
    int pidx = ck * CHUNK_SZ + row;
    int s_eff = pidx < L_TOT ? pidx : pidx - PADN;
    int t = sortedT[hbase + s_eff];
    for (int c = 0; c < 16; ++c) qL[i * 16 + c] = (_Float16)xe[(size_t)t * 16 + c];
  }
  __syncthreads();

  int lane = tid & 31; int wv = tid >> 5;
  int m = lane & 15; int hi = lane >> 4;
  int mtile = mb * 3 + wv;

  v8h z8 = {};
  v16h z16 = {};
  v16h aQ = (hi == 0) ? *(const v16h*)&qL[(wv * 16 + m) * 16] : z16;

  float rm[8], rsum[8];
  v8f o[4] = {{}, {}, {}, {}};
  for (int r = 0; r < 8; ++r) { rm[r] = -1e30f; rsum[r] = 0.f; }

  for (int nt = 0; nt < 27; ++nt) {
    // stage V tile transposed into LDS (shared by all 3 waves)
    for (int j = tid; j < 128; j += 96) {
      int row = j >> 3, seg = j & 7;
      v8h val = *(const v8h*)&yef[(size_t)gT[nt * 16 + row] * 64 + seg * 8];
      for (int c = 0; c < 8; ++c) vTt[(seg * 8 + c) * 16 + row] = val[c];
    }
    __syncthreads();

    v16h bK = (hi == 0) ? *(const v16h*)&knL[(nt * 16 + m) * 16] : z16;
    v8f zc = {};
    v8f s = wmma_f16(aQ, bK, zc);

    float tm[8];
    for (int r = 0; r < 8; ++r) tm[r] = s[r];
    for (int off = 1; off < 16; off <<= 1)
      for (int r = 0; r < 8; ++r)
        tm[r] = fmaxf(tm[r], __shfl_xor(tm[r], off, 16));

    float pv[8], ts[8];
    for (int r = 0; r < 8; ++r) {
      float nm = fmaxf(rm[r], tm[r]);
      float sc = __expf(rm[r] - nm);
      for (int ct = 0; ct < 4; ++ct) o[ct][r] *= sc;
      rsum[r] *= sc;
      pv[r] = __expf(s[r] - nm);
      ts[r] = pv[r];
      rm[r] = nm;
    }
    for (int off = 1; off < 16; off <<= 1)
      for (int r = 0; r < 8; ++r)
        ts[r] += __shfl_xor(ts[r], off, 16);
    for (int r = 0; r < 8; ++r) rsum[r] += ts[r];

    // stage P (D-layout -> A-layout) via LDS
    for (int r = 0; r < 8; ++r) st[wv][r + 8 * hi][m] = (_Float16)pv[r];
    __syncthreads();

    v16h aP = (hi == 0) ? *(const v16h*)&st[wv][m][0] : z16;
    for (int ct = 0; ct < 4; ++ct) {
      v16h bV = (hi == 0) ? *(const v16h*)&vTt[(ct * 16 + m) * 16] : z16;
      o[ct] = wmma_f16(aP, bV, o[ct]);
    }
    __syncthreads();
  }

  // scatter directly to original (h, t) positions (fuses undo_sort)
  for (int r = 0; r < 8; ++r) {
    int row = mtile * 16 + r + 8 * hi;
    int sp = ck * CHUNK_SZ + row;
    if (sp < L_TOT) {
      int t = gT[row];
      float inv = 1.f / rsum[r];
      for (int ct = 0; ct < 4; ++ct)
        retW[(hbase + t) * 64 + ct * 16 + m] = o[ct][r] * inv;
      if (m == 0) bsW[hbase + t] = rm[r] + __logf(rsum[r]);
    }
  }
}

// ---------------- K6: hash-softmax recombination + residual ---------------
__global__ void __launch_bounds__(256) k_combine(const float* __restrict__ x,
    const float* __restrict__ retW, const float* __restrict__ bsW,
    float* __restrict__ attW, _Float16* __restrict__ rattW) {
  int t = blockIdx.x * 256 + threadIdx.x;
  float bv[4]; float mx = -1e30f;
  for (int h = 0; h < 4; ++h) { bv[h] = bsW[(size_t)h * L_TOT + t]; mx = fmaxf(mx, bv[h]); }
  float sum = 0.f;
  for (int h = 0; h < 4; ++h) { bv[h] = __expf(bv[h] - mx); sum += bv[h]; }
  float invs = 1.f / sum;
  for (int h = 0; h < 4; ++h) bv[h] *= invs;
  for (int c = 0; c < 64; ++c) {
    float acc = 0.f;
    for (int h = 0; h < 4; ++h)
      acc += retW[((size_t)h * L_TOT + t) * 64 + c] * bv[h];
    float a = acc + x[(size_t)c * L_TOT + t];
    attW[(size_t)c * L_TOT + t] = a;
    rattW[(size_t)t * 64 + c] = (_Float16)fmaxf(a, 0.f);   // channel-minor for K7
  }
}

// ---------------- K7: final conv 64->64, 3x3x3 + residual -----------------
// One wave: 32 consecutive positions (2 M-tiles) x 64 output channels.
// B fragments reused across both M-tiles: 1.5 b128 loads per WMMA.
__global__ void __launch_bounds__(32) k_conv_final(const _Float16* __restrict__ rattW,
    const _Float16* __restrict__ wcB, const float* __restrict__ bc,
    const float* __restrict__ attW, float* __restrict__ out) {
  int tbase = blockIdx.x * 32;
  int lane = threadIdx.x; int m = lane & 15; int hi = lane >> 4;
  int pos0 = tbase + m;
  int d = pos0 >> 12; int h = (pos0 >> 6) & 63; int w0 = pos0 & 63;
  v8f acc[4][2] = {{{}, {}}, {{}, {}}, {{}, {}}, {{}, {}}};
  for (int tap = 0; tap < 27; ++tap) {
    int dz = tap / 9 - 1, dy = (tap / 3) % 3 - 1, dx = tap % 3 - 1;
    bool okdh = (unsigned)(d + dz) < DD && (unsigned)(h + dy) < HH;
    bool ok0 = okdh && (unsigned)(w0 + dx) < WW;
    bool ok1 = okdh && (unsigned)(w0 + 16 + dx) < WW;
    int off = dz * 4096 + dy * 64 + dx;
    int p0 = ok0 ? (pos0 + off) : pos0;
    int p1 = ok1 ? (pos0 + 16 + off) : (pos0 + 16);
    if (tap < 26) __builtin_prefetch(wcB + (size_t)(tap + 1) * 4096, 0, 1);
    for (int cbi = 0; cbi < 2; ++cbi) {
      v16h bfrag[4];
      for (int ct = 0; ct < 4; ++ct)
        bfrag[ct] = loadB32(wcB + (size_t)(((tap * 2 + cbi) * 4 + ct) * 16 + m) * 32, hi);
      v16h a0 = loadA64(rattW + (size_t)p0 * 64 + cbi * 32, hi, ok0);
      for (int ct = 0; ct < 4; ++ct) acc[ct][0] = wmma_f16(a0, bfrag[ct], acc[ct][0]);
      v16h a1 = loadA64(rattW + (size_t)p1 * 64 + cbi * 32, hi, ok1);
      for (int ct = 0; ct < 4; ++ct) acc[ct][1] = wmma_f16(a1, bfrag[ct], acc[ct][1]);
    }
  }
  for (int ct = 0; ct < 4; ++ct) {
    int c = ct * 16 + m;
    for (int mt = 0; mt < 2; ++mt)
      for (int r = 0; r < 8; ++r) {
        int p = tbase + mt * 16 + r + 8 * hi;
        out[(size_t)c * L_TOT + p] = attW[(size_t)c * L_TOT + p] + acc[ct][mt][r] + bc[c];
      }
  }
}

// ---------------- host launcher -------------------------------------------
extern "C" void kernel_launch(void* const* d_in, const int* in_sizes, int n_in,
                              void* d_out, int out_size, void* d_ws, size_t ws_size,
                              hipStream_t stream) {
  const float* x   = (const float*)d_in[0];
  const float* wm  = (const float*)d_in[1];
  const float* bm  = (const float*)d_in[2];
  const float* wa  = (const float*)d_in[3];
  const float* ba  = (const float*)d_in[4];
  const float* rot = (const float*)d_in[5];
  const float* wc  = (const float*)d_in[6];
  const float* bc  = (const float*)d_in[7];
  float* out = (float*)d_out;

  char* w = (char*)d_ws;
  size_t off = 0;
  auto alloc = [&](size_t bytes) -> void* {
    void* p = w + off;
    off = (off + bytes + 255) & ~(size_t)255;
    return p;
  };
  _Float16* xh    = (_Float16*)alloc((size_t)64 * L_TOT * 2);
  float*    xe    = (float*)   alloc((size_t)L_TOT * 16 * 4);
  _Float16* yef   = (_Float16*)alloc((size_t)L_TOT * 64 * 2);
  _Float16* wmB   = (_Float16*)alloc(27 * 2 * 16 * 32 * 2);
  _Float16* waB   = (_Float16*)alloc(2 * 4 * 16 * 32 * 2);
  _Float16* wcB   = (_Float16*)alloc(27 * 2 * 4 * 16 * 32 * 2);
  int*      codes = (int*)alloc((size_t)4 * L_TOT * 4);
  int*      bh    = (int*)alloc(4 * 64 * 128 * 4);
  int*      bo    = (int*)alloc(4 * 64 * 128 * 4);
  int*      sortedT = (int*)alloc((size_t)4 * L_TOT * 4);
  float*    retW  = (float*)alloc((size_t)4 * L_TOT * 64 * 4);
  float*    bsW   = (float*)alloc((size_t)4 * L_TOT * 4);
  float*    attW  = (float*)alloc((size_t)64 * L_TOT * 4);
  _Float16* rattW = (_Float16*)alloc((size_t)64 * L_TOT * 2);

  k_prep<<<16384, 256, 0, stream>>>(x, wm, wa, wc, xh, wmB, waB, wcB);
  k_conv_match<<<2048, 32, 0, stream>>>(xh, wmB, bm, xe);
  k_conv_assembly<<<4096, 32, 0, stream>>>(xh, waB, ba, yef);
  k_hash<<<256, 256, 0, stream>>>(xe, rot, codes);
  k_hist<<<256, 256, 0, stream>>>(codes, bh);
  k_offsets<<<4, 128, 0, stream>>>(bh, bo);
  k_sort_scatter<<<256, 1024, 0, stream>>>(codes, bo, sortedT);
  k_attention<<<NHH * KC * 3, 96, 0, stream>>>(xe, yef, sortedT, retW, bsW);
  k_combine<<<256, 256, 0, stream>>>(x, retW, bsW, attW, rattW);
  k_conv_final<<<2048, 32, 0, stream>>>(rattW, wcB, bc, attW, out);
}
